// EnhancedGCN42_78683800862986
// MI455X (gfx1250) — matrix operand order repaired and checked by reading, not compiled, not run on hardware
//
#include <hip/hip_runtime.h>
#include <stdint.h>

// ---------------------------------------------------------------------------
// CDNA5 (gfx1250) wave32 WMMA types
// ---------------------------------------------------------------------------
typedef __attribute__((ext_vector_type(16))) __bf16 v16bf;
typedef __attribute__((ext_vector_type(8)))  float  v8f;

__device__ __forceinline__ __bf16 f2bf(float f) {
    // round-to-nearest-even fp32 -> bf16
    unsigned u = __builtin_bit_cast(unsigned, f);
    unsigned r = u + 0x7FFFu + ((u >> 16) & 1u);
    unsigned short s = (unsigned short)(r >> 16);
    return __builtin_bit_cast(__bf16, s);
}
__device__ __forceinline__ __bf16 bf_zero() {
    return __builtin_bit_cast(__bf16, (unsigned short)0);
}

// ---------------------------------------------------------------------------
// Degree / symmetric-normalization weights
// ---------------------------------------------------------------------------
__global__ void k_deg_init(float* __restrict__ deg, int N) {
    int i = blockIdx.x * blockDim.x + threadIdx.x;
    if (i < N) deg[i] = 1.0f;                       // self loop contributes 1
}
__global__ void k_deg_acc(const int* __restrict__ dst, float* __restrict__ deg, int E) {
    int e = blockIdx.x * blockDim.x + threadIdx.x;
    if (e < E) atomicAdd(&deg[dst[e]], 1.0f);
}
__global__ void k_deg_rsqrt(float* __restrict__ deg, int N) {
    int i = blockIdx.x * blockDim.x + threadIdx.x;
    if (i < N) deg[i] = rsqrtf(deg[i]);             // deg >= 1 always
}

// ---------------------------------------------------------------------------
// fp32 x -> bf16 copy (initial feature conversion)
// ---------------------------------------------------------------------------
__global__ void k_f32_to_bf16(const float* __restrict__ in, __bf16* __restrict__ out,
                              size_t total) {
    size_t i = (size_t)blockIdx.x * blockDim.x + threadIdx.x;
    if (i < total) out[i] = f2bf(in[i]);
}

// ---------------------------------------------------------------------------
// Weight pre-swizzle into WMMA B-fragment order (zero-padded), so the GEMM
// B-load is a single branch-free, fully coalesced 32B load per lane.
// Layout: Wswz[((ct*nkt + kt)*32 + lane)*16 + j2], where per B-fragment lane:
//   n = lane&15, hh = lane>>4, j = j2>>1, K = 2j + 16*hh + 32*kt + (j2&1),
//   col = 16*ct + n  (contiguous k order, same as the A fragment)
// ---------------------------------------------------------------------------
__global__ void k_wswz(const float* __restrict__ W, __bf16* __restrict__ out,
                       int din, int dout, int nkt, size_t total) {
    size_t idx = (size_t)blockIdx.x * blockDim.x + threadIdx.x;
    if (idx >= total) return;
    int j2   = (int)(idx & 15);
    int lane = (int)((idx >> 4) & 31);
    int t    = (int)(idx >> 9);                      // ct*nkt + kt
    int kt   = t % nkt;
    int ct   = t / nkt;
    int hh   = lane >> 4;
    int n    = lane & 15;
    int j    = j2 >> 1;
    int kb   = 2 * j + 16 * hh + (kt << 5) + (j2 & 1);
    int col  = ct * 16 + n;
    float v  = (kb < din && col < dout) ? W[(size_t)kb * dout + col] : 0.0f;
    out[idx] = f2bf(v);
}

// ---------------------------------------------------------------------------
// WMMA GEMM:  Y[N,dout] = epi( Xb[N,din](bf16) @ W(pre-swizzled bf16) )
//   epi 0: raw fp32     (GCN layers; bias added post-aggregation)
//   epi 1: +bias fp32   (final classifier layer)
//   epi 2: bn(relu(+bias)) -> bf16  (classifier hidden layers)
// A tile (16 rows x Kpad bf16) staged to LDS with CDNA5 async LDS loads.
// ---------------------------------------------------------------------------
__global__ void k_gemm_wmma(const __bf16* __restrict__ Xb, const __bf16* __restrict__ Wswz,
                            void* __restrict__ Y, int N, int din, int dout,
                            int Kpad, int nkt, int epi, int obf,
                            const float* __restrict__ bias,
                            const float* __restrict__ gg, const float* __restrict__ bb,
                            const float* __restrict__ rm, const float* __restrict__ rv)
{
    extern __shared__ __bf16 ldsA[];                 // [16][Kpad] bf16, row-major
    const int tid  = threadIdx.x;
    const int row0 = blockIdx.x * 16;

    // ---- async stage: 16 rows x din bf16 (global) -> LDS (ASYNCcnt path) ----
    {
        const int dw = din >> 1;                     // dwords per row (din is even)
        for (int t = tid; t < 16 * dw; t += blockDim.x) {
            int r = t / dw;
            int d = t - r * dw;
            unsigned lds_off = (unsigned)(uintptr_t)(ldsA + r * Kpad + 2 * d);
            const __bf16* gp = Xb + (size_t)(row0 + r) * din + 2 * d;
            unsigned long long ga = (unsigned long long)(uintptr_t)gp;
            asm volatile("global_load_async_to_lds_b32 %0, %1, off"
                         :: "v"(lds_off), "v"(ga) : "memory");
        }
        if (Kpad > din) {                            // zero-fill K padding
            int pad = Kpad - din;
            for (int t = tid; t < 16 * pad; t += blockDim.x) {
                int r = t / pad;
                int k = din + (t - r * pad);
                ldsA[r * Kpad + k] = bf_zero();
            }
        }
        asm volatile("s_wait_asynccnt 0x0" ::: "memory");
    }
    __syncthreads();

    const int lane  = tid & 31;
    const int wave  = tid >> 5;
    const int nwave = blockDim.x >> 5;
    const int m     = lane & 15;
    const int hh    = lane >> 4;                     // lane half
    const int nct   = (dout + 15) >> 4;

    for (int ct = wave; ct < nct; ct += nwave) {
        const int col0 = ct * 16;
        v8f acc = {};
        for (int kt = 0; kt < nkt; ++kt) {
            // A fragment: consecutive-K bf16 pairs from LDS (ds_load_b32 pairs)
            v16bf afrag;
            const int kbase = kt << 5;
#pragma unroll
            for (int j = 0; j < 8; ++j) {
                int ka = 2 * j + 16 * hh + kbase;
                afrag[2 * j]     = ldsA[m * Kpad + ka];
                afrag[2 * j + 1] = ldsA[m * Kpad + ka + 1];
            }
            // B fragment: one 32B contiguous, coalesced load per lane
            const v16bf bfrag =
                *(const v16bf*)(Wswz + (((size_t)ct * nkt + kt) * 32 + lane) * 16);
            acc = __builtin_amdgcn_wmma_f32_16x16x32_bf16(false, afrag, false, bfrag,
                                                          (short)0, acc, false, false);
        }
#pragma unroll
        for (int r = 0; r < 8; ++r) {
            int row = row0 + r + 8 * hh;
            int c   = col0 + m;
            if (row < N && c < dout) {               // only partial for dout=2 tile
                float v = acc[r];
                if (epi >= 1) v += bias[c];
                if (epi == 2) {
                    v = fmaxf(v, 0.0f);
                    v = (v - rm[c]) * rsqrtf(rv[c] + 1e-5f) * gg[c] + bb[c];
                }
                if (obf) ((__bf16*)Y)[(size_t)row * dout + c] = f2bf(v);
                else     ((float*)Y)[(size_t)row * dout + c]  = v;
            }
        }
    }
}

// ---------------------------------------------------------------------------
// out[i,c] = h[i,c] * dinv[i]^2 + bias[c]   (self-loop term + GCN bias; inits
// the aggregation buffer so the edge scatter can atomically add)
// ---------------------------------------------------------------------------
__global__ void k_selfloop_bias(const float* __restrict__ h, const float* __restrict__ dinv,
                                const float* __restrict__ bias, float* __restrict__ out,
                                int ls, size_t total)
{
    size_t idx = (size_t)blockIdx.x * blockDim.x + threadIdx.x;
    if (idx >= total) return;
    int dout = 1 << ls;
    int c = (int)idx & (dout - 1);
    size_t i = idx >> ls;
    float di = dinv[i];
    out[idx] = h[idx] * di * di + bias[c];
}

// ---------------------------------------------------------------------------
// Edge scatter: out[dst] += h[src] * dinv[src]*dinv[dst]
// 4 channels per thread (float4 gather), fp32 global atomics (L2-resident).
// ---------------------------------------------------------------------------
__global__ void k_edge_scatter(const float* __restrict__ h, const int* __restrict__ src,
                               const int* __restrict__ dst, const float* __restrict__ dinv,
                               float* __restrict__ out, int dout, int ls4,
                               size_t total, int E)
{
    size_t idx = (size_t)blockIdx.x * blockDim.x + threadIdx.x;
    if (idx >= total) return;
    size_t e = idx >> ls4;
    int c4 = ((int)idx & ((1 << ls4) - 1)) << 2;
    size_t e2 = e + 16;                              // gather lookahead prefetch
    if (e2 < (size_t)E)
        __builtin_prefetch(h + (size_t)src[e2] * dout + c4, 0, 0);
    int s = src[e], d = dst[e];
    float w = dinv[s] * dinv[d];
    float4 hv = *(const float4*)(h + (size_t)s * dout + c4);
    float* op = out + (size_t)d * dout + c4;
    atomicAdd(op + 0, hv.x * w);
    atomicAdd(op + 1, hv.y * w);
    atomicAdd(op + 2, hv.z * w);
    atomicAdd(op + 3, hv.w * w);
}

// ---------------------------------------------------------------------------
// BatchNorm (eval) + ReLU, fp32 in -> bf16 out (feeds the next WMMA GEMM)
// ---------------------------------------------------------------------------
__global__ void k_bn_relu_bf16(const float* __restrict__ in, __bf16* __restrict__ out,
                               const float* __restrict__ gg, const float* __restrict__ bb,
                               const float* __restrict__ rm, const float* __restrict__ rv,
                               int ls, size_t total)
{
    size_t idx = (size_t)blockIdx.x * blockDim.x + threadIdx.x;
    if (idx >= total) return;
    int c = (int)idx & ((1 << ls) - 1);
    float v = in[idx];
    v = (v - rm[c]) * rsqrtf(rv[c] + 1e-5f) * gg[c] + bb[c];
    out[idx] = f2bf(fmaxf(v, 0.0f));
}

// ---------------------------------------------------------------------------
// Host-side orchestration
// ---------------------------------------------------------------------------
extern "C" void kernel_launch(void* const* d_in, const int* in_sizes, int n_in,
                              void* d_out, int out_size, void* d_ws, size_t ws_size,
                              hipStream_t stream)
{
    const float* x  = (const float*)d_in[0];
    const int*   ei = (const int*)d_in[1];
    const int N = in_sizes[0] / 42;
    const int E = in_sizes[1] / 2;
    const int* src = ei;
    const int* dst = ei + E;

    // ws layout: dinv[N] | bufH f32[N*256] | bufA f32[N*256] | bufXb bf16[N*256] | Wswz
    size_t bigF  = (size_t)N * 256 * sizeof(float);
    size_t bigB  = (size_t)N * 256 * sizeof(__bf16);
    size_t hOff  = (((size_t)N * sizeof(float)) + 255) & ~(size_t)255;
    size_t aOff  = hOff + bigF;
    size_t xbOff = aOff + bigF;
    size_t wOff  = xbOff + bigB;
    size_t wBytes = (size_t)256 * 256 * sizeof(__bf16);   // max Kpad*Npad
    if (ws_size < wOff + wBytes) return;
    float*  dinv  = (float*)d_ws;
    float*  bufH  = (float*)((char*)d_ws + hOff);
    float*  bufA  = (float*)((char*)d_ws + aOff);
    __bf16* bufXb = (__bf16*)((char*)d_ws + xbOff);
    __bf16* bufHb = (__bf16*)bufH;                  // bf16 alias, used in classifier only
    __bf16* wswz  = (__bf16*)((char*)d_ws + wOff);

    const int B = 256;
    k_deg_init <<<(N + B - 1) / B, B, 0, stream>>>(dinv, N);
    k_deg_acc  <<<(E + B - 1) / B, B, 0, stream>>>(dst, dinv, E);
    k_deg_rsqrt<<<(N + B - 1) / B, B, 0, stream>>>(dinv, N);

    // x -> bf16 (layer-1 GEMM input lives in bufXb as [N,42])
    {
        size_t tot = (size_t)N * 42;
        k_f32_to_bf16<<<(unsigned)((tot + B - 1) / B), B, 0, stream>>>(x, bufXb, tot);
    }

    const int gblocks = (N + 15) / 16;
    struct L { int iW, din, dout; };
    const L layers[4] = { {2, 42, 128}, {8, 128, 256}, {14, 256, 128}, {20, 128, 64} };

    auto run_gemm = [&](const __bf16* Xb, int iW, int din, int dout, int epi, int obf,
                        void* Y) {
        const float* Wm = (const float*)d_in[iW];
        int Kpad = (din + 31) & ~31;
        int nkt  = Kpad >> 5;
        int nct  = (dout + 15) >> 4;
        size_t wtot = (size_t)nct * nkt * 512;
        k_wswz<<<(unsigned)((wtot + B - 1) / B), B, 0, stream>>>(Wm, wswz, din, dout, nkt, wtot);
        int blk = (dout >= 128) ? 256 : 128;
        const float* bi = (epi >= 1) ? (const float*)d_in[iW + 1] : nullptr;
        const float* gg = (epi == 2) ? (const float*)d_in[iW + 2] : nullptr;
        const float* bb = (epi == 2) ? (const float*)d_in[iW + 3] : nullptr;
        const float* rm = (epi == 2) ? (const float*)d_in[iW + 4] : nullptr;
        const float* rv = (epi == 2) ? (const float*)d_in[iW + 5] : nullptr;
        k_gemm_wmma<<<gblocks, blk, (size_t)16 * Kpad * sizeof(__bf16), stream>>>(
            Xb, wswz, Y, N, din, dout, Kpad, nkt, epi, obf, bi, gg, bb, rm, rv);
    };

    for (int l = 0; l < 4; ++l) {
        const int din = layers[l].din, dout = layers[l].dout, iW = layers[l].iW;
        run_gemm(bufXb, iW, din, dout, /*epi*/0, /*obf*/0, bufH);

        const float* bi = (const float*)d_in[iW + 1];
        const int ls = __builtin_ctz(dout);
        const size_t tot = (size_t)N << ls;
        k_selfloop_bias<<<(unsigned)((tot + B - 1) / B), B, 0, stream>>>(
            bufH, dinv, bi, bufA, ls, tot);

        const int ls4 = ls - 2;
        const size_t etot = (size_t)E << ls4;
        k_edge_scatter<<<(unsigned)((etot + B - 1) / B), B, 0, stream>>>(
            bufH, src, dst, dinv, bufA, dout, ls4, etot, E);

        k_bn_relu_bf16<<<(unsigned)((tot + B - 1) / B), B, 0, stream>>>(
            bufA, bufXb,
            (const float*)d_in[iW + 2], (const float*)d_in[iW + 3],
            (const float*)d_in[iW + 4], (const float*)d_in[iW + 5], ls, tot);
    }

    // classifier: Linear->ReLU->BN (x2, bf16 intermediates), then Linear -> d_out
    run_gemm(bufXb, 26, 64, 64, /*epi*/2, /*obf*/1, bufHb);    // c1
    run_gemm(bufHb, 32, 64, 32, /*epi*/2, /*obf*/1, bufXb);    // c2
    run_gemm(bufXb, 38, 32,  2, /*epi*/1, /*obf*/0, d_out);    // c3
}
